// oceanGCNLSTM_45664092291474
// MI455X (gfx1250) — matrix operand order repaired and checked by reading, not compiled, not run on hardware
//
#include <hip/hip_runtime.h>
#include <hip/hip_bf16.h>

typedef __bf16 bf16;
typedef __attribute__((ext_vector_type(8)))  bf16  bf16x8;
typedef __attribute__((ext_vector_type(16))) bf16  bf16x16;
typedef __attribute__((ext_vector_type(8)))  float f32x8;
typedef __attribute__((ext_vector_type(4)))  int   v4i;

#define T_DIM 64
#define N_DIM 1024
#define H_DIM 256
#define F_RAW 1026
#define F_PAD 1056      // K padded to multiple of 32 for 16x16x32 WMMA
#define G_DIM 1024      // 4*H

#if defined(__has_builtin)
#if __has_builtin(__builtin_amdgcn_global_load_async_to_lds_b128)
#define HAVE_ASYNC_LDS 1
#endif
#if __has_builtin(__builtin_amdgcn_s_wait_asynccnt)
#define HAVE_WAIT_ASYNC 1
#endif
#endif

__device__ __forceinline__ void wait_async0() {
#if defined(HAVE_ASYNC_LDS)
#if defined(HAVE_WAIT_ASYNC)
    __builtin_amdgcn_s_wait_asynccnt(0);
#else
    asm volatile("s_wait_asynccnt 0x0" ::: "memory");
#endif
#endif
}

__device__ __forceinline__ void async_copy16(const void* g, void* l) {
#if defined(HAVE_ASYNC_LDS)
    __builtin_amdgcn_global_load_async_to_lds_b128(
        (__attribute__((address_space(1))) v4i*)g,
        (__attribute__((address_space(3))) v4i*)l,
        0, 0);
#endif
}

// ---------------------------------------------------------------------------
// deg^{-1/2} per row: one wave32 per row of Ahat = max(bin(A), I)
// ---------------------------------------------------------------------------
__launch_bounds__(256)
__global__ void dinv_kernel(const float* __restrict__ A, float* __restrict__ dinv) {
    const int wid  = threadIdx.x >> 5;
    const int lane = threadIdx.x & 31;
    const long long row = (long long)blockIdx.x * 8 + wid;      // t*N + i
    const int i = (int)(row % N_DIM);
    const float* a = A + row * N_DIM;
    int cnt = 0;
    for (int j = lane; j < N_DIM; j += 32)
        cnt += ((a[j] != 0.f) || (j == i)) ? 1 : 0;
    for (int off = 16; off; off >>= 1) cnt += __shfl_xor(cnt, off, 32);
    if (lane == 0) dinv[row] = rsqrtf((float)cnt);              // deg >= 1 (self loop)
}

// ---------------------------------------------------------------------------
// Anorm[t,i,j] = dinv_i * Ahat * dinv_j  ->  bf16 row-major
// ---------------------------------------------------------------------------
__launch_bounds__(256)
__global__ void anorm_kernel(const float* __restrict__ A, const float* __restrict__ dinv,
                             bf16* __restrict__ out) {
    const long long row = blockIdx.x;               // t*N + i
    const int i = (int)(row % N_DIM);
    const long long t = row / N_DIM;
    const float di = dinv[row];
    const float* dv = dinv + t * N_DIM;
    const float* a  = A + row * N_DIM;
    bf16* o = out + row * N_DIM;
    for (int j = threadIdx.x; j < N_DIM; j += 256) {
        float v = ((a[j] != 0.f) || (j == i)) ? di * dv[j] : 0.f;
        o[j] = (bf16)v;
    }
}

// ---------------------------------------------------------------------------
// Xfeat = concat(Xhat, anchor) padded to F_PAD, bf16
// ---------------------------------------------------------------------------
__launch_bounds__(256)
__global__ void xfeat_kernel(const float* __restrict__ Xhat, const float* __restrict__ anchor,
                             bf16* __restrict__ out) {
    const long long idx = (long long)blockIdx.x * 256 + threadIdx.x;
    if (idx >= (long long)T_DIM * N_DIM * F_PAD) return;
    const int j = (int)(idx % F_PAD);
    const long long tn = idx / F_PAD;
    float v = 0.f;
    if (j < N_DIM)      v = Xhat[tn * N_DIM + j];
    else if (j < F_RAW) v = anchor[tn * 2 + (j - N_DIM)];
    out[idx] = (bf16)v;
}

// ---------------------------------------------------------------------------
// f32 -> bf16 pack with optional transpose and zero padding
// ---------------------------------------------------------------------------
__launch_bounds__(256)
__global__ void pack_bf16(const float* __restrict__ in, bf16* __restrict__ out,
                          int outR, int outC, int inR, int inC, int tr) {
    const long long idx = (long long)blockIdx.x * 256 + threadIdx.x;
    if (idx >= (long long)outR * outC) return;
    const int r = (int)(idx / outC), c = (int)(idx % outC);
    float v = 0.f;
    if (tr) { if (c < inR && r < inC) v = in[(long long)c * inC + r]; }
    else    { if (r < inR && c < inC) v = in[(long long)r * inC + c]; }
    out[idx] = (bf16)v;
}

// ---------------------------------------------------------------------------
// Tiled bf16 WMMA GEMM:  C[M,Nc] = A[M,K] x B[K,Nc]  (+bias, +relu)
// Block = 256 threads = 8 waves.  Tile 128x128, K-step 32.
// Wave grid 4(M) x 2(N); each wave owns 2x4 tiles of 16x16 (f32 accum).
// A tile staged via GLOBAL_LOAD_ASYNC_TO_LDS_B128 (ASYNCcnt) when available;
// B tile transposed into LDS [n][k] so every lane's fragment is two
// contiguous 16B reads matching the WMMA VGPR striping.
// M, Nc multiples of 128; K multiple of 32 (guaranteed by padding).
// ---------------------------------------------------------------------------
__launch_bounds__(256)
__global__ void gemm_bf16_wmma(const bf16* __restrict__ A, long long sA, int lda,
                               const bf16* __restrict__ B, long long sB, int ldb,
                               bf16* __restrict__ Cb, float* __restrict__ Cf,
                               long long sC, int ldc,
                               int K, const float* __restrict__ bias, int relu) {
    const int tid = threadIdx.x;
    const int bz  = blockIdx.z;
    A += (long long)bz * sA;
    B += (long long)bz * sB;
    const long long cbase = (long long)bz * sC;
    const int tileM = blockIdx.y * 128, tileN = blockIdx.x * 128;

    __shared__ __align__(16) bf16 As[128 * 32];
    __shared__ __align__(16) bf16 Bs[128 * 32];   // transposed: [n][k]

    const int wid = tid >> 5, lane = tid & 31;
    const int waveM = (wid & 3) * 32, waveN = (wid >> 2) * 64;
    const int lhalf = lane >> 4, l15 = lane & 15;

    f32x8 acc[2][4];
#pragma unroll
    for (int mt = 0; mt < 2; ++mt)
#pragma unroll
        for (int nt = 0; nt < 4; ++nt) acc[mt][nt] = (f32x8){};

    const int rowh = tid >> 2;          // 0..63
    const int kc   = (tid & 3) * 8;     // 0,8,16,24

    for (int kk = 0; kk < K; kk += 32) {
        // ---- stage A tile (128x32): async direct-to-LDS when available
#if defined(HAVE_ASYNC_LDS)
#pragma unroll
        for (int it = 0; it < 2; ++it) {
            const int row = it * 64 + rowh;
            async_copy16(&A[(long long)(tileM + row) * lda + kk + kc],
                         &As[row * 32 + kc]);
        }
#else
#pragma unroll
        for (int it = 0; it < 2; ++it) {
            const int row = it * 64 + rowh;
            *(bf16x8*)&As[row * 32 + kc] =
                *(const bf16x8*)&A[(long long)(tileM + row) * lda + kk + kc];
        }
#endif
        // ---- stage B tile (32x128) transposed into Bs[n][k]
#pragma unroll
        for (int it = 0; it < 2; ++it) {
            const int chunk = it * 256 + tid;      // 0..511
            const int krow  = chunk >> 4;          // 0..31
            const int n0    = (chunk & 15) * 8;
            bf16x8 v = *(const bf16x8*)&B[(long long)(kk + krow) * ldb + tileN + n0];
#pragma unroll
            for (int e = 0; e < 8; ++e) Bs[(n0 + e) * 32 + krow] = v[e];
        }
        // ---- prefetch next K tile into caches (global_prefetch_b8)
        if (kk + 32 < K) {
            __builtin_prefetch(&A[(long long)(tileM + rowh) * lda + kk + 32 + kc], 0, 1);
            __builtin_prefetch(&B[(long long)(kk + 32 + (tid >> 4)) * ldb + tileN + (tid & 15) * 8], 0, 1);
        }
        wait_async0();
        __syncthreads();

        // fragments per ISA layout: lanes 0-15 -> K[0:8)+K[16:24) (A and B)
        bf16x16 afr[2], bfr[4];
#pragma unroll
        for (int mt = 0; mt < 2; ++mt) {
            const int row = waveM + mt * 16 + l15;
            bf16x8 lo = *(const bf16x8*)&As[row * 32 + lhalf * 8];
            bf16x8 hi = *(const bf16x8*)&As[row * 32 + lhalf * 8 + 16];
            afr[mt] = __builtin_shufflevector(lo, hi, 0,1,2,3,4,5,6,7,8,9,10,11,12,13,14,15);
        }
#pragma unroll
        for (int nt = 0; nt < 4; ++nt) {
            const int n = waveN + nt * 16 + l15;
            bf16x8 lo = *(const bf16x8*)&Bs[n * 32 + lhalf * 8];
            bf16x8 hi = *(const bf16x8*)&Bs[n * 32 + lhalf * 8 + 16];
            bfr[nt] = __builtin_shufflevector(lo, hi, 0,1,2,3,4,5,6,7,8,9,10,11,12,13,14,15);
        }
#pragma unroll
        for (int mt = 0; mt < 2; ++mt)
#pragma unroll
            for (int nt = 0; nt < 4; ++nt)
                acc[mt][nt] = __builtin_amdgcn_wmma_f32_16x16x32_bf16(
                    false, afr[mt], false, bfr[nt],
                    (short)0, acc[mt][nt], false, false);
        __syncthreads();
    }

    // epilogue: C/D layout -> row = 8*(lane>=16) + vgpr, col = lane&15
#pragma unroll
    for (int mt = 0; mt < 2; ++mt) {
#pragma unroll
        for (int nt = 0; nt < 4; ++nt) {
#pragma unroll
            for (int p = 0; p < 8; ++p) {
                const int row = tileM + waveM + mt * 16 + lhalf * 8 + p;
                const int col = tileN + waveN + nt * 16 + l15;
                float v = acc[mt][nt][p];
                if (bias) v += bias[col];
                if (relu) v = v > 0.f ? v : 0.f;
                if (Cf) Cf[cbase + (long long)row * ldc + col] = v;
                else    Cb[cbase + (long long)row * ldc + col] = (bf16)v;
            }
        }
    }
}

// ---------------------------------------------------------------------------
// LSTM elementwise cell (gate order i,f,g,o); Gx precomputed for all t
// ---------------------------------------------------------------------------
__launch_bounds__(256)
__global__ void lstm_init(bf16* __restrict__ h, float* __restrict__ c) {
    const int idx = blockIdx.x * 256 + threadIdx.x;
    h[idx] = (bf16)0.f;
    c[idx] = 0.f;
}

__launch_bounds__(256)
__global__ void lstm_cell(const bf16* __restrict__ gx, const float* __restrict__ gh,
                          const float* __restrict__ bih, const float* __restrict__ bhh,
                          bf16* __restrict__ h, float* __restrict__ c,
                          bf16* __restrict__ hs, int t) {
    const int idx = blockIdx.x * 256 + threadIdx.x;   // n*H + k
    const int n = idx >> 8, k = idx & 255;
    const long long gxb = ((long long)t * N_DIM + n) * G_DIM;
    const long long ghb = (long long)n * G_DIM;
    const float gi = (float)gx[gxb + k]       + gh[ghb + k]       + bih[k]       + bhh[k];
    const float gf = (float)gx[gxb + 256 + k] + gh[ghb + 256 + k] + bih[256 + k] + bhh[256 + k];
    const float gg = (float)gx[gxb + 512 + k] + gh[ghb + 512 + k] + bih[512 + k] + bhh[512 + k];
    const float go = (float)gx[gxb + 768 + k] + gh[ghb + 768 + k] + bih[768 + k] + bhh[768 + k];
    const float si = 1.f / (1.f + __expf(-gi));
    const float sf = 1.f / (1.f + __expf(-gf));
    const float so = 1.f / (1.f + __expf(-go));
    const float cn = sf * c[idx] + si * tanhf(gg);
    const float hn = so * tanhf(cn);
    c[idx] = cn;
    h[idx] = (bf16)hn;
    hs[(long long)t * N_DIM * H_DIM + idx] = (bf16)hn;
}

// ---------------------------------------------------------------------------
// Final FC: out[t,n,:] = hs[t,n,:] @ Wfc + bfc   (O=2, tiny)
// ---------------------------------------------------------------------------
__launch_bounds__(256)
__global__ void fc_kernel(const bf16* __restrict__ hs, const float* __restrict__ Wfc,
                          const float* __restrict__ bfc, float* __restrict__ out) {
    const long long tn = (long long)blockIdx.x * 256 + threadIdx.x;
    const bf16* hrow = hs + tn * H_DIM;
    float a0 = 0.f, a1 = 0.f;
    for (int k = 0; k < H_DIM; ++k) {
        const float hv = (float)hrow[k];
        a0 += hv * Wfc[k * 2 + 0];
        a1 += hv * Wfc[k * 2 + 1];
    }
    out[tn * 2 + 0] = a0 + bfc[0];
    out[tn * 2 + 1] = a1 + bfc[1];
}

// ---------------------------------------------------------------------------
extern "C" void kernel_launch(void* const* d_in, const int* in_sizes, int n_in,
                              void* d_out, int out_size, void* d_ws, size_t ws_size,
                              hipStream_t stream) {
    (void)in_sizes; (void)n_in; (void)out_size; (void)ws_size;
    const float* Xhat   = (const float*)d_in[0];
    const float* Aadj   = (const float*)d_in[1];
    const float* anchor = (const float*)d_in[2];
    const float* W1 = (const float*)d_in[3];
    const float* b1 = (const float*)d_in[4];
    const float* W2 = (const float*)d_in[5];
    const float* b2 = (const float*)d_in[6];
    const float* W3 = (const float*)d_in[7];
    const float* b3 = (const float*)d_in[8];
    const float* Wih = (const float*)d_in[9];
    const float* Whh = (const float*)d_in[10];
    const float* bih = (const float*)d_in[11];
    const float* bhh = (const float*)d_in[12];
    const float* Wfc = (const float*)d_in[13];
    const float* bfc = (const float*)d_in[14];
    float* out = (float*)d_out;

    char* base = (char*)d_ws;
    size_t off = 0;
    auto carve = [&](size_t bytes) -> char* {
        char* p = base + off;
        off = (off + bytes + 255) & ~(size_t)255;
        return p;
    };

    const long long TN = (long long)T_DIM * N_DIM;
    bf16*  anorm = (bf16*)carve(TN * N_DIM * sizeof(bf16));     // 128 MB
    bf16*  gx    = anorm;                                       // aliased after GCN phase
    bf16*  xfeat = (bf16*)carve(TN * F_PAD * sizeof(bf16));     // 132 MB
    bf16*  zbuf  = (bf16*)carve(TN * H_DIM * sizeof(bf16));
    bf16*  xa    = (bf16*)carve(TN * H_DIM * sizeof(bf16));
    bf16*  xb    = (bf16*)carve(TN * H_DIM * sizeof(bf16));
    bf16*  hsbuf = (bf16*)carve(TN * H_DIM * sizeof(bf16));
    bf16*  w1p   = (bf16*)carve((size_t)F_PAD * H_DIM * sizeof(bf16));
    bf16*  w2b   = (bf16*)carve((size_t)H_DIM * H_DIM * sizeof(bf16));
    bf16*  w3b   = (bf16*)carve((size_t)H_DIM * H_DIM * sizeof(bf16));
    bf16*  wihT  = (bf16*)carve((size_t)H_DIM * G_DIM * sizeof(bf16));
    bf16*  whhT  = (bf16*)carve((size_t)H_DIM * G_DIM * sizeof(bf16));
    float* dinv  = (float*)carve(TN * sizeof(float));
    float* gh    = (float*)carve((size_t)N_DIM * G_DIM * sizeof(float));
    bf16*  hbuf  = (bf16*)carve((size_t)N_DIM * H_DIM * sizeof(bf16));
    float* cbuf  = (float*)carve((size_t)N_DIM * H_DIM * sizeof(float));

    // ---- graph normalization + feature/weight packing
    dinv_kernel <<<(int)(TN / 8), 256, 0, stream>>>(Aadj, dinv);
    anorm_kernel<<<(int)TN, 256, 0, stream>>>(Aadj, dinv, anorm);
    xfeat_kernel<<<(int)((TN * F_PAD + 255) / 256), 256, 0, stream>>>(Xhat, anchor, xfeat);
    pack_bf16<<<(F_PAD * H_DIM + 255) / 256, 256, 0, stream>>>(W1, w1p, F_PAD, H_DIM, F_RAW, H_DIM, 0);
    pack_bf16<<<(H_DIM * H_DIM + 255) / 256, 256, 0, stream>>>(W2, w2b, H_DIM, H_DIM, H_DIM, H_DIM, 0);
    pack_bf16<<<(H_DIM * H_DIM + 255) / 256, 256, 0, stream>>>(W3, w3b, H_DIM, H_DIM, H_DIM, H_DIM, 0);
    pack_bf16<<<(H_DIM * G_DIM + 255) / 256, 256, 0, stream>>>(Wih, wihT, H_DIM, G_DIM, G_DIM, H_DIM, 1);
    pack_bf16<<<(H_DIM * G_DIM + 255) / 256, 256, 0, stream>>>(Whh, whhT, H_DIM, G_DIM, G_DIM, H_DIM, 1);

    const long long sNN = (long long)N_DIM * N_DIM;
    const long long sNH = (long long)N_DIM * H_DIM;

    // ---- GCN layer 1:  Z = Xfeat @ W1p ; X = relu(Anorm @ Z + b1)
    gemm_bf16_wmma<<<dim3(H_DIM / 128, (int)(TN / 128), 1), 256, 0, stream>>>(
        xfeat, 0, F_PAD, w1p, 0, H_DIM, zbuf, nullptr, 0, H_DIM, F_PAD, nullptr, 0);
    gemm_bf16_wmma<<<dim3(H_DIM / 128, N_DIM / 128, T_DIM), 256, 0, stream>>>(
        anorm, sNN, N_DIM, zbuf, sNH, H_DIM, xa, nullptr, sNH, H_DIM, N_DIM, b1, 1);
    // ---- GCN layer 2
    gemm_bf16_wmma<<<dim3(H_DIM / 128, (int)(TN / 128), 1), 256, 0, stream>>>(
        xa, 0, H_DIM, w2b, 0, H_DIM, zbuf, nullptr, 0, H_DIM, H_DIM, nullptr, 0);
    gemm_bf16_wmma<<<dim3(H_DIM / 128, N_DIM / 128, T_DIM), 256, 0, stream>>>(
        anorm, sNN, N_DIM, zbuf, sNH, H_DIM, xb, nullptr, sNH, H_DIM, N_DIM, b2, 1);
    // ---- GCN layer 3
    gemm_bf16_wmma<<<dim3(H_DIM / 128, (int)(TN / 128), 1), 256, 0, stream>>>(
        xb, 0, H_DIM, w3b, 0, H_DIM, zbuf, nullptr, 0, H_DIM, H_DIM, nullptr, 0);
    gemm_bf16_wmma<<<dim3(H_DIM / 128, N_DIM / 128, T_DIM), 256, 0, stream>>>(
        anorm, sNN, N_DIM, zbuf, sNH, H_DIM, xa, nullptr, sNH, H_DIM, N_DIM, b3, 1);

    // ---- LSTM input gates for all timesteps at once: Gx = X3 @ Wih^T
    // (gx aliases anorm: Anorm is dead beyond this point, stream order guarantees it)
    gemm_bf16_wmma<<<dim3(G_DIM / 128, (int)(TN / 128), 1), 256, 0, stream>>>(
        xa, 0, H_DIM, wihT, 0, G_DIM, gx, nullptr, 0, G_DIM, H_DIM, nullptr, 0);

    // ---- recurrence
    lstm_init<<<(N_DIM * H_DIM) / 256, 256, 0, stream>>>(hbuf, cbuf);
    for (int t = 0; t < T_DIM; ++t) {
        gemm_bf16_wmma<<<dim3(G_DIM / 128, N_DIM / 128, 1), 256, 0, stream>>>(
            hbuf, 0, H_DIM, whhT, 0, G_DIM, nullptr, gh, 0, G_DIM, H_DIM, nullptr, 0);
        lstm_cell<<<(N_DIM * H_DIM) / 256, 256, 0, stream>>>(
            gx, gh, bih, bhh, hbuf, cbuf, hsbuf, t);
    }

    // ---- final projection
    fc_kernel<<<(int)(TN / 256), 256, 0, stream>>>(hsbuf, Wfc, bfc, out);
}
